// Attention_27762668601471
// MI455X (gfx1250) — compile-verified, matrix operand order checked by reading, not run on hardware
//
#include <hip/hip_runtime.h>

#define SEQ 2048
#define DMODEL 1024
#define NHEAD 16
#define DK 64
#define LN_EPS 1e-5f

typedef __bf16 v16bf  __attribute__((ext_vector_type(16)));
typedef __bf16 bf16x8 __attribute__((ext_vector_type(8)));
typedef __bf16 bf16x4 __attribute__((ext_vector_type(4)));
typedef float  v8f    __attribute__((ext_vector_type(8)));
typedef float  f32x4  __attribute__((ext_vector_type(4)));
typedef int    v4i    __attribute__((ext_vector_type(4)));

#define AS1 __attribute__((address_space(1)))
#define AS3 __attribute__((address_space(3)))

#if __has_builtin(__builtin_amdgcn_global_load_async_to_lds_b128) && \
    __has_builtin(__builtin_amdgcn_s_wait_asynccnt)
#define HAVE_ASYNC_LDS 1
#else
#define HAVE_ASYNC_LDS 0
#endif

// K-index mapping for 16-bit A/B fragments of V_WMMA_*_16X16X32_* (ISA 7.12.2):
// element e -> K = {0..7}+8*hl for e<8, {16..23}+8*hl for e>=8.
__device__ __forceinline__ int kmap(int e, int hl) {
    int v = e >> 1;
    int base = (v < 4) ? (2 * v) : (16 + 2 * (v - 4));
    return base + (e & 1) + 8 * hl;
}

// Fragment whose element e reads p[kmap(e,hl)]: two contiguous 16B runs.
__device__ __forceinline__ v16bf load_frag(const __bf16* p, int hl) {
    bf16x8 a0 = *(const bf16x8*)(p + 8 * hl);
    bf16x8 a1 = *(const bf16x8*)(p + 16 + 8 * hl);
    return __builtin_shufflevector(a0, a1, 0, 1, 2, 3, 4, 5, 6, 7,
                                   8, 9, 10, 11, 12, 13, 14, 15);
}

// ---------------------------------------------------------------------------
// Tiled bf16 WMMA GEMM: out[N,Dout] = A[N,K] @ W[K,Dout] + bias (+ resid)
// 256 threads = 8 waves (2x4), block tile 128x128, wave tile 64x32, K-step 32.
// MODE 0: A = f32, store bf16.  MODE 1: A = bf16, store f32 + bias + residual.
// ---------------------------------------------------------------------------
template <int MODE>
__global__ __launch_bounds__(256)
void gemm_wmma(const float* __restrict__ A, const __bf16* __restrict__ Abf,
               const float* __restrict__ W, const float* __restrict__ bias,
               __bf16* __restrict__ outbf, float* __restrict__ outf,
               const float* __restrict__ resid, int N, int K, int Dout) {
    __shared__ __align__(16) __bf16 lA[128 * 32];
    __shared__ __align__(16) __bf16 lB[32 * 128];

    const int tid  = threadIdx.x;
    const int wave = tid >> 5;
    const int lane = tid & 31;
    const int lm   = lane & 15;
    const int hl   = lane >> 4;
    const int wrow = wave >> 2;            // 0..1  -> 64-row slab
    const int wcol = wave & 3;             // 0..3  -> 32-col slab
    const int row0 = blockIdx.y * 128;
    const int col0 = blockIdx.x * 128;

    v8f zero = {};
    v8f acc[4][2];
#pragma unroll
    for (int i = 0; i < 4; ++i)
#pragma unroll
        for (int j = 0; j < 2; ++j) acc[i][j] = zero;

    for (int k0 = 0; k0 < K; k0 += 32) {
        // ---- stage A tile (128x32) ----
#pragma unroll
        for (int rep = 0; rep < 4; ++rep) {
            int idx = tid + rep * 256;                 // 0..1023 vec4 chunks
            int r = idx >> 3, c4 = (idx & 7) * 4;
            if constexpr (MODE == 0) {
                f32x4 v = *(const f32x4*)&A[(size_t)(row0 + r) * K + k0 + c4];
                bf16x4 b = {(__bf16)v.x, (__bf16)v.y, (__bf16)v.z, (__bf16)v.w};
                *(bf16x4*)&lA[r * 32 + c4] = b;
            } else {
                *(bf16x4*)&lA[r * 32 + c4] =
                    *(const bf16x4*)&Abf[(size_t)(row0 + r) * K + k0 + c4];
            }
        }
        // ---- stage B tile (32x128) ----
#pragma unroll
        for (int rep = 0; rep < 4; ++rep) {
            int idx = tid + rep * 256;
            int r = idx >> 5, c4 = (idx & 31) * 4;
            f32x4 v = *(const f32x4*)&W[(size_t)(k0 + r) * Dout + col0 + c4];
            bf16x4 b = {(__bf16)v.x, (__bf16)v.y, (__bf16)v.z, (__bf16)v.w};
            *(bf16x4*)&lB[r * 128 + c4] = b;
        }
        __syncthreads();

        v16bf af[4], bfr[2];
#pragma unroll
        for (int i = 0; i < 4; ++i)
            af[i] = load_frag(&lA[(wrow * 64 + i * 16 + lm) * 32], hl);
#pragma unroll
        for (int j = 0; j < 2; ++j) {
            int n0 = wcol * 32 + j * 16 + lm;
#pragma unroll
            for (int e = 0; e < 16; ++e)              // K-stride 128: scattered
                bfr[j][e] = lB[kmap(e, hl) * 128 + n0];
        }
#pragma unroll
        for (int i = 0; i < 4; ++i)
#pragma unroll
            for (int j = 0; j < 2; ++j)
                acc[i][j] = __builtin_amdgcn_wmma_f32_16x16x32_bf16(
                    false, af[i], false, bfr[j], (short)0, acc[i][j], false, false);
        __syncthreads();
    }

#pragma unroll
    for (int i = 0; i < 4; ++i)
#pragma unroll
        for (int j = 0; j < 2; ++j) {
            int gcol = col0 + wcol * 32 + j * 16 + lm;
            float bv = bias[gcol];
#pragma unroll
            for (int v = 0; v < 8; ++v) {
                int grow = row0 + wrow * 64 + i * 16 + v + 8 * hl;
                size_t gi = (size_t)grow * Dout + gcol;
                float val = acc[i][j][v] + bv;
                if constexpr (MODE == 0) outbf[gi] = (__bf16)val;
                else                     outf[gi] = val + resid[gi];
            }
        }
}

// ---------------------------------------------------------------------------
// Flash attention: grid = (SEQ/64, NHEAD), 128 thr (4 waves), wave = 16 q rows.
// K/V tiles staged in LDS once per block (K via async-to-LDS when available,
// V transposed so PV B-fragments vectorize). P re-striped via per-wave LDS.
// ---------------------------------------------------------------------------
__global__ __launch_bounds__(128)
void attn_flash(const __bf16* __restrict__ Q, const __bf16* __restrict__ Kb,
                const __bf16* __restrict__ Vb, __bf16* __restrict__ ctx) {
    __shared__ __align__(16) __bf16 lQ[64 * 64];       // 8 KB
    __shared__ __align__(16) __bf16 lK[32 * 64];       // 4 KB  [key][d]
    __shared__ __align__(16) __bf16 lVt[64 * 32];      // 4 KB  [d][key]
    __shared__ __align__(16) __bf16 lP[4][16 * 32];    // 4 KB

    const int tid   = threadIdx.x;
    const int wave  = tid >> 5;
    const int lane  = tid & 31;
    const int lm    = lane & 15;
    const int hl    = lane >> 4;
    const int qtile = blockIdx.x;
    const int h     = blockIdx.y;
    const int qbase = qtile * 64 + wave * 16;
    const int hoff  = h * DK;

    // stage Q tile (64 rows x 64 dk) once
#pragma unroll
    for (int rep = 0; rep < 8; ++rep) {
        int idx = tid + rep * 128;                     // 0..1023 vec4 chunks
        int r = idx >> 4, c4 = (idx & 15) * 4;
        *(bf16x4*)&lQ[r * 64 + c4] =
            *(const bf16x4*)&Q[(size_t)(qtile * 64 + r) * DMODEL + hoff + c4];
    }
    __syncthreads();
    v16bf qf[2];
#pragma unroll
    for (int c = 0; c < 2; ++c)
        qf[c] = load_frag(&lQ[(wave * 16 + lm) * 64 + c * 32], hl);

    v8f zero = {};
    v8f acc[4];
#pragma unroll
    for (int t = 0; t < 4; ++t) acc[t] = zero;
    float mrun[8], lrun[8];
#pragma unroll
    for (int v = 0; v < 8; ++v) { mrun[v] = -__builtin_inff(); lrun[v] = 0.f; }

    const int nkb = qtile * 2 + 2;                      // causal bound
    for (int kb = 0; kb < nkb; ++kb) {
        const int kbase = kb * 32;

        if (kb + 1 < nkb) {                             // prefetch next K/V rows
            __builtin_prefetch(&Kb[(size_t)(kbase + 32 + lane) * DMODEL + hoff], 0, 1);
            __builtin_prefetch(&Vb[(size_t)(kbase + 32 + lane) * DMODEL + hoff], 0, 1);
        }
        __syncthreads();                                // previous tile reads done

        // ---- stage K tile (32 keys x 64 d) ----
#if HAVE_ASYNC_LDS
#pragma unroll
        for (int rep = 0; rep < 2; ++rep) {
            int idx = tid + rep * 128;                  // 0..255 x 16B
            int r = idx >> 3, c8 = (idx & 7) * 8;
            __builtin_amdgcn_global_load_async_to_lds_b128(
                (AS1 v4i*)&Kb[(size_t)(kbase + r) * DMODEL + hoff + c8],
                (AS3 v4i*)&lK[r * 64 + c8], 0, 0);
        }
#else
#pragma unroll
        for (int rep = 0; rep < 4; ++rep) {
            int idx = tid + rep * 128;
            int r = idx >> 4, c4 = (idx & 15) * 4;
            *(bf16x4*)&lK[r * 64 + c4] =
                *(const bf16x4*)&Kb[(size_t)(kbase + r) * DMODEL + hoff + c4];
        }
#endif
        // ---- stage V tile transposed: lVt[d][key] ----
#pragma unroll
        for (int rep = 0; rep < 4; ++rep) {
            int idx = tid + rep * 128;
            int r = idx >> 4, c4 = (idx & 15) * 4;      // key r, dims c4..c4+3
            bf16x4 v = *(const bf16x4*)&Vb[(size_t)(kbase + r) * DMODEL + hoff + c4];
#pragma unroll
            for (int k = 0; k < 4; ++k) lVt[(c4 + k) * 32 + r] = v[k];
        }
#if HAVE_ASYNC_LDS
        __builtin_amdgcn_s_wait_asynccnt(0);
#endif
        __syncthreads();

        // ---- S = Q @ K^T ----
        v8f s[2];
        s[0] = zero; s[1] = zero;
#pragma unroll
        for (int t = 0; t < 2; ++t)
#pragma unroll
            for (int c = 0; c < 2; ++c) {
                v16bf kf = load_frag(&lK[(t * 16 + lm) * 64 + c * 32], hl);
                s[t] = __builtin_amdgcn_wmma_f32_16x16x32_bf16(
                    false, qf[c], false, kf, (short)0, s[t], false, false);
            }

        // ---- scale + causal mask + row max ----
        float rmax[8];
#pragma unroll
        for (int v = 0; v < 8; ++v) {
            int q = qbase + v + 8 * hl;
            float m = -__builtin_inff();
#pragma unroll
            for (int t = 0; t < 2; ++t) {
                int key = kbase + t * 16 + lm;
                float sv = s[t][v] * 0.125f;
                sv = (key > q) ? -__builtin_inff() : sv;
                s[t][v] = sv;
                m = fmaxf(m, sv);
            }
            rmax[v] = m;
        }
#pragma unroll
        for (int off = 1; off < 16; off <<= 1)
#pragma unroll
            for (int v = 0; v < 8; ++v)
                rmax[v] = fmaxf(rmax[v], __shfl_xor(rmax[v], off, 32));

        // ---- online softmax ----
        float alpha[8];
#pragma unroll
        for (int v = 0; v < 8; ++v) {
            float mnew = fmaxf(mrun[v], rmax[v]);
            alpha[v] = __expf(mrun[v] - mnew);
            mrun[v] = mnew;
        }
        float rsum[8];
#pragma unroll
        for (int v = 0; v < 8; ++v) {
            float sum = 0.f;
#pragma unroll
            for (int t = 0; t < 2; ++t) {
                float p = __expf(s[t][v] - mrun[v]);
                s[t][v] = p;
                sum += p;
            }
            rsum[v] = sum;
        }
#pragma unroll
        for (int off = 1; off < 16; off <<= 1)
#pragma unroll
            for (int v = 0; v < 8; ++v)
                rsum[v] += __shfl_xor(rsum[v], off, 32);
#pragma unroll
        for (int v = 0; v < 8; ++v) lrun[v] = lrun[v] * alpha[v] + rsum[v];
#pragma unroll
        for (int t = 0; t < 4; ++t)
#pragma unroll
            for (int v = 0; v < 8; ++v) acc[t][v] *= alpha[v];

        // ---- P: C layout -> A layout via per-wave LDS tile ----
#pragma unroll
        for (int t = 0; t < 2; ++t)
#pragma unroll
            for (int v = 0; v < 8; ++v)
                lP[wave][(v + 8 * hl) * 32 + t * 16 + lm] = (__bf16)s[t][v];
        __syncthreads();                                // uniform trip count
        v16bf pf = load_frag(&lP[wave][lm * 32], hl);

        // ---- ctx += P @ V ----
#pragma unroll
        for (int t = 0; t < 4; ++t) {
            v16bf vf = load_frag(&lVt[(t * 16 + lm) * 32], hl);
            acc[t] = __builtin_amdgcn_wmma_f32_16x16x32_bf16(
                false, pf, false, vf, (short)0, acc[t], false, false);
        }
    }

#pragma unroll
    for (int t = 0; t < 4; ++t)
#pragma unroll
        for (int v = 0; v < 8; ++v) {
            int qrow = qbase + v + 8 * hl;
            float o = acc[t][v] / lrun[v];
            ctx[(size_t)qrow * DMODEL + hoff + t * 16 + lm] = (__bf16)o;
        }
}

// ---------------------------------------------------------------------------
// Row-wise LayerNorm, in place on d_out.
// ---------------------------------------------------------------------------
__global__ __launch_bounds__(256)
void layernorm_kernel(float* __restrict__ y, const float* __restrict__ gamma,
                      const float* __restrict__ beta) {
    __shared__ float ssum[8], ssum2[8];
    const int row = blockIdx.x;
    const int tid = threadIdx.x;
    float s = 0.f, s2 = 0.f;
    for (int i = tid; i < DMODEL; i += 256) {
        float v = y[(size_t)row * DMODEL + i];
        s += v; s2 += v * v;
    }
#pragma unroll
    for (int off = 1; off < 32; off <<= 1) {
        s  += __shfl_xor(s,  off, 32);
        s2 += __shfl_xor(s2, off, 32);
    }
    int wave = tid >> 5, lane = tid & 31;
    if (lane == 0) { ssum[wave] = s; ssum2[wave] = s2; }
    __syncthreads();
    float ts = 0.f, ts2 = 0.f;
#pragma unroll
    for (int w = 0; w < 8; ++w) { ts += ssum[w]; ts2 += ssum2[w]; }
    float mu  = ts * (1.f / DMODEL);
    float var = ts2 * (1.f / DMODEL) - mu * mu;
    float inv = rsqrtf(var + LN_EPS);
    for (int i = tid; i < DMODEL; i += 256) {
        float v = y[(size_t)row * DMODEL + i];
        y[(size_t)row * DMODEL + i] = (v - mu) * inv * gamma[i] + beta[i];
    }
}

extern "C" void kernel_launch(void* const* d_in, const int* in_sizes, int n_in,
                              void* d_out, int out_size, void* d_ws, size_t ws_size,
                              hipStream_t stream) {
    const float* x     = (const float*)d_in[0];
    const float* Wq    = (const float*)d_in[1];
    const float* bq    = (const float*)d_in[2];
    const float* Wk    = (const float*)d_in[3];
    const float* bk    = (const float*)d_in[4];
    const float* Wv    = (const float*)d_in[5];
    const float* bv    = (const float*)d_in[6];
    const float* Wo    = (const float*)d_in[7];
    const float* bo    = (const float*)d_in[8];
    const float* gamma = (const float*)d_in[9];
    const float* beta  = (const float*)d_in[10];
    float* out = (float*)d_out;

    char* ws = (char*)d_ws;
    const size_t MAT = (size_t)SEQ * DMODEL * sizeof(__bf16);  // 4 MB
    __bf16* Qb = (__bf16*)(ws);
    __bf16* Kb = (__bf16*)(ws + MAT);
    __bf16* Vb = (__bf16*)(ws + 2 * MAT);
    __bf16* Cb = (__bf16*)(ws + 3 * MAT);

    dim3 gGemm(DMODEL / 128, SEQ / 128);
    gemm_wmma<0><<<gGemm, 256, 0, stream>>>(x, nullptr, Wq, bq, Qb, nullptr,
                                            nullptr, SEQ, DMODEL, DMODEL);
    gemm_wmma<0><<<gGemm, 256, 0, stream>>>(x, nullptr, Wk, bk, Kb, nullptr,
                                            nullptr, SEQ, DMODEL, DMODEL);
    gemm_wmma<0><<<gGemm, 256, 0, stream>>>(x, nullptr, Wv, bv, Vb, nullptr,
                                            nullptr, SEQ, DMODEL, DMODEL);
    dim3 gAttn(SEQ / 64, NHEAD);
    attn_flash<<<gAttn, 128, 0, stream>>>(Qb, Kb, Vb, Cb);
    gemm_wmma<1><<<gGemm, 256, 0, stream>>>(nullptr, Cb, Wo, bo, nullptr, out, x,
                                            SEQ, DMODEL, DMODEL);
    layernorm_kernel<<<SEQ, 256, 0, stream>>>(out, gamma, beta);
}
